// Attention_36086315221474
// MI455X (gfx1250) — matrix operand order repaired and checked by reading, not compiled, not run on hardware
//
#include <hip/hip_runtime.h>
#include <hip/hip_bf16.h>

// MHA forward for B=8, N=1024, DIM=1024, H=16, D=64 on gfx1250.
// All matmuls via v_wmma_f32_16x16x32_f16 (f32 accumulate).
// Flash-style attention (S^T orientation) -> no O(N^2) materialization.
// GEMMs use 64x64 wave tiles (4x4 WMMA tiles): 16 b128 loads per 16 WMMAs.
// Workspace usage: ~88 MB (f16 staging of x/weights/Q/K/V^T/attn-out + mask bias).

#define B_   8
#define N_   1024
#define DIM_ 1024
#define H_   16
#define D_   64
// SCALE = DIM^-0.5 = 1/32; fold log2(e) so softmax runs in exp2 domain.
#define QSCALE (0.03125f * 1.44269504088896f)
#define NEGBIG (-1.0e30f)

typedef _Float16 half8_t  __attribute__((ext_vector_type(8)));
typedef _Float16 half16_t __attribute__((ext_vector_type(16)));
typedef float    float8_t __attribute__((ext_vector_type(8)));

union Frag {
  half16_t v;
  half8_t  h[2];
  unsigned u[8];
};

__device__ __forceinline__ float8_t zero8() {
  float8_t z = {0.f, 0.f, 0.f, 0.f, 0.f, 0.f, 0.f, 0.f};
  return z;
}

__device__ __forceinline__ float8_t wmma16(const Frag& a, const Frag& b, float8_t c) {
  return __builtin_amdgcn_wmma_f32_16x16x32_f16(false, a.v, false, b.v,
                                                (short)0, c, false, false);
}

// A fragment (16x32 f16, MxK): lane holds row L%16; one contiguous 16-half
// run at k0 + 16*(L/16).  `p` must already include
// row*ld + 16*(lane>>4); load at +k0 and +k0+8.
__device__ __forceinline__ Frag load_a_at(const _Float16* p, int k0) {
  Frag f;
  f.h[0] = *(const half8_t*)(p + k0);
  f.h[1] = *(const half8_t*)(p + k0 + 8);
  return f;
}

// B fragment (32x16 f16, KxN) from a row-major matrix M where B[k][n] =
// M[row_n][k]: lane holds column L%16; one contiguous 16-half run at
// k0 + 16*(lane>>4).  `p` must already include row*ld + 16*(lane>>4).
__device__ __forceinline__ Frag load_b_at(const _Float16* p, int k0) {
  Frag f;
  f.h[0] = *(const half8_t*)(p + k0);
  f.h[1] = *(const half8_t*)(p + k0 + 8);
  return f;
}

__device__ __forceinline__ Frag load_a(const _Float16* base, int row, int ld,
                                       int k0, int lane) {
  return load_a_at(base + (size_t)row * ld + ((lane >> 4) << 4), k0);
}
__device__ __forceinline__ Frag load_b(const _Float16* base, int row, int ld,
                                       int k0, int lane) {
  return load_b_at(base + (size_t)row * ld + ((lane >> 4) << 4), k0);
}

// SWAPX16 lane exchange via ds_swizzle (group-of-32, xor=0x10, and=0x1f).
__device__ __forceinline__ float swz16f(float x) {
  return __int_as_float(__builtin_amdgcn_ds_swizzle(__float_as_int(x), 0x401F));
}
__device__ __forceinline__ unsigned swz16u(unsigned x) {
  return (unsigned)__builtin_amdgcn_ds_swizzle((int)x, 0x401F);
}
__device__ __forceinline__ unsigned pack2h(float a, float b) {
  union { _Float16 h[2]; unsigned u; } z;
  z.h[0] = (_Float16)a; z.h[1] = (_Float16)b;
  return z.u;
}
__device__ __forceinline__ float fexp2(float x) {
  return __builtin_amdgcn_exp2f(x);
}

// ---------------------------------------------------------------------------
__global__ void cvt_f32_f16(const float* __restrict__ src,
                            _Float16* __restrict__ dst, int n) {
  int i = blockIdx.x * blockDim.x + threadIdx.x;
  int stride = gridDim.x * blockDim.x;
  for (; i < n; i += stride) dst[i] = (_Float16)src[i];
}

// mask: bool [B, N-1] (1 byte each). Row 0 (CLS) always visible.
__global__ void build_maskbias(const unsigned char* __restrict__ mask,
                               float* __restrict__ mb) {
  int i = blockIdx.x * blockDim.x + threadIdx.x;
  if (i >= B_ * N_) return;
  int b = i >> 10, n = i & (N_ - 1);
  mb[i] = (n == 0) ? 0.f : (mask[b * (N_ - 1) + n - 1] ? 0.f : NEGBIG);
}

// ---------------------------------------------------------------------------
// Fused QKV projection: [8192,1024] x_f16 @ W^T -> 3072 features.
// j<1024 -> Q (+pos, *QSCALE) to qh[b,h,n,d]; 1024..2047 -> K (+pos) to
// kh[b,h,n,d]; >=2048 -> V to vth[b,h,d,n] (transposed for the PV gemm).
// Wave tile 64x64 (4x4 WMMA tiles); block = 4 waves (2x2) -> 128x128 per WG.
__global__ __launch_bounds__(128) void gemm_qkv(
    const _Float16* __restrict__ xh, const _Float16* __restrict__ wqk,
    const _Float16* __restrict__ wv, const float* __restrict__ pos,
    _Float16* __restrict__ qh, _Float16* __restrict__ kh,
    _Float16* __restrict__ vth) {
  const int tid = threadIdx.x, lane = tid & 31, wave = tid >> 5;
  const int m0 = blockIdx.x * 128 + (wave >> 1) * 64;
  const int n0 = blockIdx.y * 128 + (wave & 1) * 64;

  // Hoist per-tile base pointers (lane k-group offset folded in).
  const _Float16* aptr[4];
  const _Float16* bptr[4];
#pragma unroll
  for (int i = 0; i < 4; ++i)
    aptr[i] = xh + (size_t)(m0 + i * 16 + (lane & 15)) * DIM_ +
              ((lane >> 4) << 4);
#pragma unroll
  for (int j = 0; j < 4; ++j) {
    const int ct = n0 + j * 16;                 // tile col base (wave-uniform)
    const int c = ct + (lane & 15);
    const _Float16* wp =
        (ct < 2048) ? (wqk + (size_t)c * DIM_) : (wv + (size_t)(c - 2048) * DIM_);
    bptr[j] = wp + ((lane >> 4) << 4);
  }

  float8_t acc[4][4];
#pragma unroll
  for (int i = 0; i < 4; ++i)
#pragma unroll
    for (int j = 0; j < 4; ++j) acc[i][j] = zero8();

  for (int k0 = 0; k0 < DIM_; k0 += 32) {
    Frag a[4], bf[4];
#pragma unroll
    for (int i = 0; i < 4; ++i) a[i] = load_a_at(aptr[i], k0);
#pragma unroll
    for (int j = 0; j < 4; ++j) bf[j] = load_b_at(bptr[j], k0);
#pragma unroll
    for (int i = 0; i < 4; ++i)
#pragma unroll
      for (int j = 0; j < 4; ++j) acc[i][j] = wmma16(a[i], bf[j], acc[i][j]);
  }

#pragma unroll
  for (int i = 0; i < 4; ++i) {
    const int rowb = m0 + i * 16 + ((lane >> 4) << 3);  // first of 8 rows
    const int bb = rowb >> 10;                          // batch
    const int nb = rowb & (N_ - 1);                     // token (contig in r)
#pragma unroll
    for (int j = 0; j < 4; ++j) {
      const int ct = n0 + j * 16;
      const int c = ct + (lane & 15);
      if (ct < 1024) {  // Q: add pos, fold SCALE*log2e
        const int h = c >> 6, d = c & 63;
        _Float16* dst = qh + ((size_t)(bb * H_ + h) * N_) * D_ + d;
#pragma unroll
        for (int r = 0; r < 8; ++r) {
          float v = (acc[i][j][r] + pos[(size_t)(rowb + r) * DIM_ + c]) * QSCALE;
          dst[(size_t)(nb + r) * D_] = (_Float16)v;
        }
      } else if (ct < 2048) {  // K: add pos
        const int cc = c - 1024;
        const int h = cc >> 6, d = cc & 63;
        _Float16* dst = kh + ((size_t)(bb * H_ + h) * N_) * D_ + d;
#pragma unroll
        for (int r = 0; r < 8; ++r) {
          float v = acc[i][j][r] + pos[(size_t)(rowb + r) * DIM_ + cc];
          dst[(size_t)(nb + r) * D_] = (_Float16)v;
        }
      } else {  // V -> transposed [bh, d, n]; 8 contiguous n per lane
        const int cc = c - 2048;
        const int h = cc >> 6, d = cc & 63;
        half8_t hv;
#pragma unroll
        for (int r = 0; r < 8; ++r) hv[r] = (_Float16)acc[i][j][r];
        *(half8_t*)(vth + ((size_t)(bb * H_ + h) * D_ + d) * N_ + nb) = hv;
      }
    }
  }
}

// ---------------------------------------------------------------------------
// Flash attention. One WG (8 waves) = one (b,h) x 128-query block; each wave
// owns 16 queries. S^T = K*Q^T per 16x16 tile puts each query's kv-scores in
// one lane (column = query); online softmax is per-lane with a lane^16 joint
// max. exp(S^T) tiles convert in-register (one swizzle per pair) into the
// B-fragment of O^T += V^T * P^T.
__global__ __launch_bounds__(256) void flash_attn(
    const _Float16* __restrict__ qh, const _Float16* __restrict__ kh,
    const _Float16* __restrict__ vth, const float* __restrict__ mb,
    _Float16* __restrict__ ao) {
  const int tid = threadIdx.x, lane = tid & 31, wave = tid >> 5;
  const int bh = blockIdx.x >> 3;
  const int b = bh >> 4, h = bh & 15;
  const int q0 = (blockIdx.x & 7) * 128 + wave * 16;

  __shared__ float smask[N_];
  for (int i = tid; i < N_; i += 256) smask[i] = mb[b * N_ + i];
  __syncthreads();

  const _Float16* Qb = qh + (size_t)bh * N_ * D_;
  const _Float16* Kb = kh + (size_t)bh * N_ * D_;
  const _Float16* Vb = vth + (size_t)bh * D_ * N_;

  const int qrow = q0 + (lane & 15);
  const int kgrp = (lane >> 4) << 3;
  Frag qf0 = load_b(Qb, qrow, D_, 0, lane);   // B-frag of Q^T, d 0..31
  Frag qf1 = load_b(Qb, qrow, D_, 32, lane);  // d 32..63
  const bool rowvalid = (smask[qrow] == 0.0f);

  float8_t ot[4];
#pragma unroll
  for (int t = 0; t < 4; ++t) ot[t] = zero8();
  float m_run = NEGBIG, l_run = 0.f;

  for (int kv0 = 0; kv0 < N_; kv0 += 32) {
    // S^T tiles: [kv 16 x q 16], two kv sub-tiles, K-dim = D = 64
    float8_t st[2];
#pragma unroll
    for (int t = 0; t < 2; ++t) {
      Frag a0 = load_a(Kb, kv0 + t * 16 + (lane & 15), D_, 0, lane);
      Frag a1 = load_a(Kb, kv0 + t * 16 + (lane & 15), D_, 32, lane);
      st[t] = wmma16(a1, qf1, wmma16(a0, qf0, zero8()));
    }
    // mask + online softmax (per-lane; joint max with lane^16 partner)
    float p[2][8];
    float mloc = m_run;
#pragma unroll
    for (int t = 0; t < 2; ++t)
#pragma unroll
      for (int r = 0; r < 8; ++r) {
        float s = rowvalid ? (st[t][r] + smask[kv0 + t * 16 + kgrp + r])
                           : NEGBIG;
        p[t][r] = s;
        mloc = fmaxf(mloc, s);
      }
    mloc = fmaxf(mloc, swz16f(mloc));
    const float rescale = fexp2(m_run - mloc);
    m_run = mloc;
    float psum = 0.f;
#pragma unroll
    for (int t = 0; t < 2; ++t)
#pragma unroll
      for (int r = 0; r < 8; ++r) {
        float e = fexp2(p[t][r] - mloc);
        p[t][r] = e;
        psum += e;
      }
    l_run = l_run * rescale + psum;
#pragma unroll
    for (int t = 0; t < 4; ++t)
#pragma unroll
      for (int r = 0; r < 8; ++r) ot[t][r] *= rescale;

    // Build P^T B-fragment (32 kv x 16 q) from the two C tiles.
    Frag bp;
    const bool lo = (lane < 16);
#pragma unroll
    for (int v = 0; v < 4; ++v) {
      unsigned u0 = pack2h(p[0][2 * v], p[0][2 * v + 1]);
      unsigned u1 = pack2h(p[1][2 * v], p[1][2 * v + 1]);
      unsigned x0 = swz16u(u0);
      unsigned x1 = swz16u(u1);
      bp.u[v]     = lo ? u0 : x1;
      bp.u[v + 4] = lo ? x0 : u1;
    }
    // O^T[d,q] += V^T[d,kv] * P^T[kv,q], 4 d-tiles
#pragma unroll
    for (int t = 0; t < 4; ++t) {
      Frag av = load_a(Vb, t * 16 + (lane & 15), N_, kv0, lane);
      ot[t] = wmma16(av, bp, ot[t]);
    }
  }

  float l_tot = l_run + swz16f(l_run);
  float inv = 1.0f / l_tot;
  // O^T lane layout: column q = L%16, d = t*16 + r + 8*(L/16) -> 16B stores
  _Float16* dst = ao + ((size_t)(b * N_) + qrow) * DIM_ + h * D_ + kgrp;
#pragma unroll
  for (int t = 0; t < 4; ++t) {
    half8_t hv;
#pragma unroll
    for (int r = 0; r < 8; ++r) hv[r] = (_Float16)(ot[t][r] * inv);
    *(half8_t*)(dst + t * 16) = hv;
  }
}

// ---------------------------------------------------------------------------
// Output projection: attn_out_f16 [8192,1024] @ W_out^T + b_out -> f32 out.
// Wave tile 64x64 (4x4), block = 4 waves (2x2) -> 128x128 per WG.
__global__ __launch_bounds__(128) void gemm_out(
    const _Float16* __restrict__ ah, const _Float16* __restrict__ wo,
    const float* __restrict__ bo, float* __restrict__ out) {
  const int tid = threadIdx.x, lane = tid & 31, wave = tid >> 5;
  const int m0 = blockIdx.x * 128 + (wave >> 1) * 64;
  const int n0 = blockIdx.y * 128 + (wave & 1) * 64;

  const _Float16* aptr[4];
  const _Float16* bptr[4];
#pragma unroll
  for (int i = 0; i < 4; ++i)
    aptr[i] = ah + (size_t)(m0 + i * 16 + (lane & 15)) * DIM_ +
              ((lane >> 4) << 4);
#pragma unroll
  for (int j = 0; j < 4; ++j)
    bptr[j] = wo + (size_t)(n0 + j * 16 + (lane & 15)) * DIM_ +
              ((lane >> 4) << 4);

  float8_t acc[4][4];
#pragma unroll
  for (int i = 0; i < 4; ++i)
#pragma unroll
    for (int j = 0; j < 4; ++j) acc[i][j] = zero8();

  for (int k0 = 0; k0 < DIM_; k0 += 32) {
    Frag a[4], bf[4];
#pragma unroll
    for (int i = 0; i < 4; ++i) a[i] = load_a_at(aptr[i], k0);
#pragma unroll
    for (int j = 0; j < 4; ++j) bf[j] = load_b_at(bptr[j], k0);
#pragma unroll
    for (int i = 0; i < 4; ++i)
#pragma unroll
      for (int j = 0; j < 4; ++j) acc[i][j] = wmma16(a[i], bf[j], acc[i][j]);
  }

#pragma unroll
  for (int i = 0; i < 4; ++i) {
    const int rowb = m0 + i * 16 + ((lane >> 4) << 3);
#pragma unroll
    for (int j = 0; j < 4; ++j) {
      const int c = n0 + j * 16 + (lane & 15);
      const float bias = bo[c];
#pragma unroll
      for (int r = 0; r < 8; ++r)
        out[(size_t)(rowb + r) * DIM_ + c] = acc[i][j][r] + bias;
    }
  }
}

// ---------------------------------------------------------------------------
extern "C" void kernel_launch(void* const* d_in, const int* in_sizes, int n_in,
                              void* d_out, int out_size, void* d_ws,
                              size_t ws_size, hipStream_t stream) {
  const float* x = (const float*)d_in[0];
  const unsigned char* mask = (const unsigned char*)d_in[1];  // jnp.bool_ = 1B
  const float* pos = (const float*)d_in[2];
  const float* Wqk = (const float*)d_in[3];
  const float* Wv = (const float*)d_in[4];
  const float* Wout = (const float*)d_in[5];
  const float* bout = (const float*)d_in[6];
  float* out = (float*)d_out;

  char* ws = (char*)d_ws;
  size_t off = 0;
  auto alloc = [&](size_t bytes) {
    void* p = ws + off;
    off += (bytes + 255) & ~(size_t)255;
    return p;
  };
  const size_t MN = (size_t)B_ * N_;          // 8192 rows
  _Float16* xh   = (_Float16*)alloc(MN * DIM_ * 2);            // 16 MB
  _Float16* wqkh = (_Float16*)alloc((size_t)2048 * DIM_ * 2);  // 4 MB
  _Float16* wvh  = (_Float16*)alloc((size_t)1024 * DIM_ * 2);  // 2 MB
  _Float16* woh  = (_Float16*)alloc((size_t)1024 * DIM_ * 2);  // 2 MB
  _Float16* qh   = (_Float16*)alloc(MN * DIM_ * 2);            // 16 MB [b,h,n,d]
  _Float16* kh   = (_Float16*)alloc(MN * DIM_ * 2);            // 16 MB [b,h,n,d]
  _Float16* vth  = (_Float16*)alloc(MN * DIM_ * 2);            // 16 MB [b,h,d,n]
  float*    mb   = (float*)alloc(MN * 4);                      // 32 KB
  _Float16* aoh  = (_Float16*)alloc(MN * DIM_ * 2);            // 16 MB
  (void)ws_size; (void)n_in; (void)in_sizes; (void)out_size;

  // Stage 0: f16 staging + mask bias
  cvt_f32_f16<<<2048, 256, 0, stream>>>(x, xh, (int)(MN * DIM_));
  cvt_f32_f16<<<1024, 256, 0, stream>>>(Wqk, wqkh, 2048 * DIM_);
  cvt_f32_f16<<<512, 256, 0, stream>>>(Wv, wvh, 1024 * DIM_);
  cvt_f32_f16<<<512, 256, 0, stream>>>(Wout, woh, 1024 * DIM_);
  build_maskbias<<<(B_ * N_ + 255) / 256, 256, 0, stream>>>(mask, mb);

  // Stage 1: fused QKV projection (+pos, scale folded into Q) -> f16 layouts
  gemm_qkv<<<dim3(64, 24), 128, 0, stream>>>(xh, wqkh, wvh, pos, qh, kh, vth);

  // Stage 2: flash attention, 1 WG per (b,h,128-query block)
  flash_attn<<<B_ * H_ * (N_ / 128), 256, 0, stream>>>(qh, kh, vth, mb, aoh);

  // Stage 3: output projection + bias -> f32
  gemm_out<<<dim3(64, 8), 128, 0, stream>>>(aoh, woh, bout, out);
}
